// OPTAttention_42709154791550
// MI455X (gfx1250) — matrix operand order repaired and checked by reading, not compiled, not run on hardware
//
#include <hip/hip_runtime.h>

#define DI __device__ __forceinline__

typedef __attribute__((ext_vector_type(16))) __bf16 v16bf;
typedef __attribute__((ext_vector_type(8)))  float  v8f;
typedef unsigned short u16;
typedef unsigned int   u32;

constexpr int B_ = 2, T_ = 2048, E_ = 2048, H_ = 32, HD_ = 64;
constexpr float SCALING = 0.125f;  // HD^-0.5 = 1/8

// 32-byte fragment: 16 bf16 values per lane (A or B operand of 16x16x32 WMMA)
union Frag { v16bf v; uint4 q[2]; };

DI u16 f2bf(float f) {
  u32 u = __float_as_uint(f);
  u32 r = (u + 0x7FFFu + ((u >> 16) & 1u)) >> 16;  // round-to-nearest-even
  return (u16)r;
}

DI v8f v8zero() {
  v8f z = {0.f, 0.f, 0.f, 0.f, 0.f, 0.f, 0.f, 0.f};
  return z;
}

DI v8f wmma_bf16(v16bf a, v16bf b, v8f c) {
  // D = A(16x32 bf16) * B(32x16 bf16) + C(16x16 f32)
  return __builtin_amdgcn_wmma_f32_16x16x32_bf16(
      /*neg_a=*/false, a, /*neg_b=*/false, b,
      /*c_mod=*/(short)0, c, /*reuse_a=*/false, /*reuse_b=*/false);
}

// ---------------------------------------------------------------------------
// f32 -> bf16 conversion, vectorized (float4 in, ushort4 out)
// ---------------------------------------------------------------------------
__global__ void to_bf16_kernel(const float4* __restrict__ in,
                               ushort4* __restrict__ out, int n4) {
  int i = blockIdx.x * blockDim.x + threadIdx.x;
  if (i < n4) {
    float4 v = in[i];
    ushort4 r;
    r.x = f2bf(v.x); r.y = f2bf(v.y); r.z = f2bf(v.z); r.w = f2bf(v.w);
    out[i] = r;
  }
}

// ---------------------------------------------------------------------------
// QKV projection: y = x @ W^T + b.  One wave computes a 64x64 output tile
// (4x4 subtiles, 16 WMMA accumulators); per K-step 8 fragment loads feed
// 16 WMMAs.  blockIdx.z: 0=Q (scaled, head-split), 1=K, 2=V (transposed).
// N-tile width 64 == HD, so each tile maps to exactly one head.
// ---------------------------------------------------------------------------
__global__ __launch_bounds__(128) void qkv_gemm_kernel(
    const u16* __restrict__ xb,
    const u16* __restrict__ Wqb, const u16* __restrict__ Wkb,
    const u16* __restrict__ Wvb,
    const float* __restrict__ bq, const float* __restrict__ bk,
    const float* __restrict__ bv,
    u16* __restrict__ Qs, u16* __restrict__ Kb, u16* __restrict__ Vt) {
  const int lane = threadIdx.x & 31;
  const int wave = threadIdx.x >> 5;
  const int l15 = lane & 15, lhi = lane >> 4;
  const int NT = E_ / 64;  // 32 col tiles

  int flat = blockIdx.x * 4 + wave;   // 2048 tiles total
  int mt = flat / NT, nt = flat % NT; // 64 x 32 tiles of 64x64
  int proj = blockIdx.z;
  const u16* W = (proj == 0) ? Wqb : (proj == 1 ? Wkb : Wvb);
  const float* bias = (proj == 0) ? bq : (proj == 1 ? bk : bv);

  // A fragment base: lane = row; lanes>=16 take K offsets +8/+24
  const u16* abase = xb + (size_t)(mt * 64 + l15) * E_ + lhi * 8;
  // B fragment base: lane = col (= row of W); lanes>=16 take K offsets +8/+24
  const u16* bbase = W + (size_t)(nt * 64 + l15) * E_ + lhi * 8;

  v8f acc[4][4];
#pragma unroll
  for (int i = 0; i < 4; ++i)
#pragma unroll
    for (int j = 0; j < 4; ++j) acc[i][j] = v8zero();

  for (int k = 0; k < E_; k += 32) {
    Frag a[4], b[4];
#pragma unroll
    for (int i = 0; i < 4; ++i) {
      const u16* ap = abase + (size_t)(i * 16) * E_ + k;
      a[i].q[0] = *(const uint4*)ap;
      a[i].q[1] = *(const uint4*)(ap + 16);
      const u16* bp = bbase + (size_t)(i * 16) * E_ + k;
      b[i].q[0] = *(const uint4*)bp;
      b[i].q[1] = *(const uint4*)(bp + 16);
    }
    if (k + 32 < E_) {
      __builtin_prefetch(abase + k + 32, 0, 1);
      __builtin_prefetch(bbase + k + 32, 0, 1);
    }
#pragma unroll
    for (int i = 0; i < 4; ++i)
#pragma unroll
      for (int j = 0; j < 4; ++j)
        acc[i][j] = wmma_bf16(a[i].v, b[j].v, acc[i][j]);
  }

  const int h = nt;  // 64-wide N tile == one head
#pragma unroll
  for (int jn = 0; jn < 4; ++jn) {
    int ncol = nt * 64 + jn * 16 + l15;  // output feature index
    int d = jn * 16 + l15;               // dim within head
    float bval = bias[ncol];
#pragma unroll
    for (int im = 0; im < 4; ++im) {
#pragma unroll
      for (int j = 0; j < 8; ++j) {
        int mrow = mt * 64 + im * 16 + j + 8 * lhi;  // global row in [0, B*T)
        int b0 = mrow / T_, t = mrow % T_;
        float val = acc[im][jn][j] + bval;
        if (proj == 0) {
          Qs[(((size_t)(b0 * H_ + h)) * T_ + t) * HD_ + d] =
              f2bf(val * SCALING);
        } else if (proj == 1) {
          Kb[(((size_t)(b0 * H_ + h)) * T_ + t) * HD_ + d] = f2bf(val);
        } else {  // V stored transposed per head: (HD, T)
          Vt[(((size_t)(b0 * H_ + h)) * HD_ + d) * T_ + t] = f2bf(val);
        }
      }
    }
  }
}

// ---------------------------------------------------------------------------
// Flash attention: one wave owns a 32-query tile (two 16-row subtiles that
// SHARE the K and V fragments of each 32-key step -> 16 WMMAs per K/V load
// set).  S = Q@K^T, causal mask, online softmax, O += P@V.  P is re-layouted
// C->A through a per-wave LDS buffer (reused sequentially; same-wave DS ops
// are in-order so WAR through LDS is safe).
// ---------------------------------------------------------------------------
__global__ __launch_bounds__(128) void flash_attn_kernel(
    const u16* __restrict__ Qs, const u16* __restrict__ Kb,
    const u16* __restrict__ Vt, u16* __restrict__ ao) {
  __shared__ u16 plds[4][16 * 32];  // 1 KB per wave

  const int lane = threadIdx.x & 31, wave = threadIdx.x >> 5;
  const int l15 = lane & 15, lhi = lane >> 4;

  int flat = blockIdx.x * 4 + wave;
  int bh = flat >> 6;  // / (T/32 = 64)
  int qt = flat & 63;
  int qb = qt * 32;

  const u16* Qh = Qs + (size_t)bh * T_ * HD_;
  const u16* Kh = Kb + (size_t)bh * T_ * HD_;
  const u16* Vh = Vt + (size_t)bh * HD_ * T_;

  // Q fragments: 2 q-subtiles x 2 K-dim steps of HD=64; live across key loop
  Frag qf[2][2];
#pragma unroll
  for (int qs = 0; qs < 2; ++qs)
#pragma unroll
    for (int ks = 0; ks < 2; ++ks) {
      const u16* p = Qh + (size_t)(qb + qs * 16 + l15) * HD_ + ks * 32 + lhi * 8;
      qf[qs][ks].q[0] = *(const uint4*)p;
      qf[qs][ks].q[1] = *(const uint4*)(p + 16);
    }

  v8f o[2][4];  // per q-subtile 16x64 output accumulator
  float m[2][8], lsum[2][8];
#pragma unroll
  for (int qs = 0; qs < 2; ++qs) {
#pragma unroll
    for (int i = 0; i < 4; ++i) o[qs][i] = v8zero();
#pragma unroll
    for (int j = 0; j < 8; ++j) { m[qs][j] = -3.0e38f; lsum[qs][j] = 0.f; }
  }

  u16* pl = plds[wave];

  for (int kb0 = 0; kb0 < qb + 32; kb0 += 32) {
    // ---- scores for both q-subtiles with shared K fragments
    v8f s0[2], s1[2];
#pragma unroll
    for (int qs = 0; qs < 2; ++qs) { s0[qs] = v8zero(); s1[qs] = v8zero(); }
#pragma unroll
    for (int ks = 0; ks < 2; ++ks) {
      Frag kb_0, kb_1;
      const u16* k0 = Kh + (size_t)(kb0 + l15) * HD_ + ks * 32 + lhi * 8;
      const u16* k1 = k0 + 16 * HD_;  // keys +16
      kb_0.q[0] = *(const uint4*)k0;  kb_0.q[1] = *(const uint4*)(k0 + 16);
      kb_1.q[0] = *(const uint4*)k1;  kb_1.q[1] = *(const uint4*)(k1 + 16);
#pragma unroll
      for (int qs = 0; qs < 2; ++qs) {
        s0[qs] = wmma_bf16(qf[qs][ks].v, kb_0.v, s0[qs]);
        s1[qs] = wmma_bf16(qf[qs][ks].v, kb_1.v, s1[qs]);
      }
    }

    // ---- V fragments (shared by both q-subtiles); V^T so cols contiguous
    Frag vb[4];
#pragma unroll
    for (int n2 = 0; n2 < 4; ++n2) {
      const u16* vp = Vh + (size_t)(n2 * 16 + l15) * T_ + kb0 + lhi * 8;
      vb[n2].q[0] = *(const uint4*)vp;
      vb[n2].q[1] = *(const uint4*)(vp + 16);
    }

#pragma unroll
    for (int qs = 0; qs < 2; ++qs) {
      // ---- causal mask + online softmax (rows split across 16-lane halves)
#pragma unroll
      for (int j = 0; j < 8; ++j) {
        int row = qb + qs * 16 + j + 8 * lhi;
        if (kb0 + l15 > row)      s0[qs][j] = -3.0e38f;
        if (kb0 + 16 + l15 > row) s1[qs][j] = -3.0e38f;
        float rmax = fmaxf(s0[qs][j], s1[qs][j]);
#pragma unroll
        for (int mk = 8; mk >= 1; mk >>= 1)
          rmax = fmaxf(rmax, __shfl_xor(rmax, mk, 32));
        float mn = fmaxf(m[qs][j], rmax);
        float sc = __expf(m[qs][j] - mn);
        float p0 = __expf(s0[qs][j] - mn);
        float p1 = __expf(s1[qs][j] - mn);
        s0[qs][j] = p0; s1[qs][j] = p1;
        float rs = p0 + p1;
#pragma unroll
        for (int mk = 8; mk >= 1; mk >>= 1)
          rs += __shfl_xor(rs, mk, 32);
        lsum[qs][j] = lsum[qs][j] * sc + rs;
        m[qs][j] = mn;
        o[qs][0][j] *= sc; o[qs][1][j] *= sc;
        o[qs][2][j] *= sc; o[qs][3][j] *= sc;
      }

      // ---- P: C-layout -> LDS -> A-layout fragment
#pragma unroll
      for (int j = 0; j < 8; ++j) {
        int row = j + 8 * lhi;
        pl[row * 32 + l15]      = f2bf(s0[qs][j]);
        pl[row * 32 + 16 + l15] = f2bf(s1[qs][j]);
      }
      Frag pf;
      const u16* pr = pl + l15 * 32 + lhi * 8;
      pf.q[0] = *(const uint4*)pr;
      pf.q[1] = *(const uint4*)(pr + 16);

      // ---- O(16x64) += P(16x32) @ V(32x64)
#pragma unroll
      for (int n2 = 0; n2 < 4; ++n2)
        o[qs][n2] = wmma_bf16(pf.v, vb[n2].v, o[qs][n2]);
    }
  }

  // ---- normalize and write attn output as bf16 in (B, T, E) layout
  int b0 = bh >> 5, h = bh & 31;  // H = 32
#pragma unroll
  for (int qs = 0; qs < 2; ++qs) {
#pragma unroll
    for (int j = 0; j < 8; ++j) {
      float inv = 1.0f / lsum[qs][j];
      int t = qb + qs * 16 + j + 8 * lhi;
      size_t base = ((size_t)(b0 * T_ + t)) * E_ + h * HD_ + l15;
      ao[base + 0]  = f2bf(o[qs][0][j] * inv);
      ao[base + 16] = f2bf(o[qs][1][j] * inv);
      ao[base + 32] = f2bf(o[qs][2][j] * inv);
      ao[base + 48] = f2bf(o[qs][3][j] * inv);
    }
  }
}

// ---------------------------------------------------------------------------
// Output projection: out = ao @ Wo^T + bo, f32 output; 64x64 tile per wave
// ---------------------------------------------------------------------------
__global__ __launch_bounds__(128) void out_proj_kernel(
    const u16* __restrict__ ao, const u16* __restrict__ Wob,
    const float* __restrict__ bo, float* __restrict__ out) {
  const int lane = threadIdx.x & 31;
  const int wave = threadIdx.x >> 5;
  const int l15 = lane & 15, lhi = lane >> 4;
  const int NT = E_ / 64;

  int flat = blockIdx.x * 4 + wave;
  int mt = flat / NT, nt = flat % NT;

  const u16* abase = ao + (size_t)(mt * 64 + l15) * E_ + lhi * 8;
  const u16* bbase = Wob + (size_t)(nt * 64 + l15) * E_ + lhi * 8;

  v8f acc[4][4];
#pragma unroll
  for (int i = 0; i < 4; ++i)
#pragma unroll
    for (int j = 0; j < 4; ++j) acc[i][j] = v8zero();

  for (int k = 0; k < E_; k += 32) {
    Frag a[4], b[4];
#pragma unroll
    for (int i = 0; i < 4; ++i) {
      const u16* ap = abase + (size_t)(i * 16) * E_ + k;
      a[i].q[0] = *(const uint4*)ap;
      a[i].q[1] = *(const uint4*)(ap + 16);
      const u16* bp = bbase + (size_t)(i * 16) * E_ + k;
      b[i].q[0] = *(const uint4*)bp;
      b[i].q[1] = *(const uint4*)(bp + 16);
    }
    if (k + 32 < E_) {
      __builtin_prefetch(abase + k + 32, 0, 1);
      __builtin_prefetch(bbase + k + 32, 0, 1);
    }
#pragma unroll
    for (int i = 0; i < 4; ++i)
#pragma unroll
      for (int j = 0; j < 4; ++j)
        acc[i][j] = wmma_bf16(a[i].v, b[j].v, acc[i][j]);
  }

#pragma unroll
  for (int jn = 0; jn < 4; ++jn) {
    int ncol = nt * 64 + jn * 16 + l15;
    float bval = bo[ncol];
#pragma unroll
    for (int im = 0; im < 4; ++im) {
#pragma unroll
      for (int j = 0; j < 8; ++j) {
        int mrow = mt * 64 + im * 16 + j + 8 * lhi;
        out[(size_t)mrow * E_ + ncol] = acc[im][jn][j] + bval;
      }
    }
  }
}

// ---------------------------------------------------------------------------
extern "C" void kernel_launch(void* const* d_in, const int* in_sizes, int n_in,
                              void* d_out, int out_size, void* d_ws,
                              size_t ws_size, hipStream_t stream) {
  const float* hs = (const float*)d_in[0];
  // d_in[1] = attention_mask (analytically causal; not needed)
  const float* Wq = (const float*)d_in[2];
  const float* bq = (const float*)d_in[3];
  const float* Wk = (const float*)d_in[4];
  const float* bk = (const float*)d_in[5];
  const float* Wv = (const float*)d_in[6];
  const float* bv = (const float*)d_in[7];
  const float* Wo = (const float*)d_in[8];
  const float* bo = (const float*)d_in[9];
  float* out = (float*)d_out;

  const size_t xe = (size_t)B_ * T_ * E_;  // 8,388,608
  const size_t ee = (size_t)E_ * E_;       // 4,194,304

  // workspace layout (bf16 elements); total ~117 MB
  u16* xb  = (u16*)d_ws;
  u16* wqb = xb + xe;
  u16* wkb = wqb + ee;
  u16* wvb = wkb + ee;
  u16* wob = wvb + ee;
  u16* Qs  = wob + ee;
  u16* Kbf = Qs + xe;
  u16* Vt  = Kbf + xe;
  u16* ao  = Vt + xe;

  const int thr = 256;
  to_bf16_kernel<<<(int)((xe / 4 + thr - 1) / thr), thr, 0, stream>>>(
      (const float4*)hs, (ushort4*)xb, (int)(xe / 4));
  to_bf16_kernel<<<(int)((ee / 4 + thr - 1) / thr), thr, 0, stream>>>(
      (const float4*)Wq, (ushort4*)wqb, (int)(ee / 4));
  to_bf16_kernel<<<(int)((ee / 4 + thr - 1) / thr), thr, 0, stream>>>(
      (const float4*)Wk, (ushort4*)wkb, (int)(ee / 4));
  to_bf16_kernel<<<(int)((ee / 4 + thr - 1) / thr), thr, 0, stream>>>(
      (const float4*)Wv, (ushort4*)wvb, (int)(ee / 4));
  to_bf16_kernel<<<(int)((ee / 4 + thr - 1) / thr), thr, 0, stream>>>(
      (const float4*)Wo, (ushort4*)wob, (int)(ee / 4));

  // QKV: (B*T/64) * (E/64) = 2048 tiles of 64x64, 4 waves/block, z = 3 proj
  dim3 gq(((B_ * T_ / 64) * (E_ / 64)) / 4, 1, 3);
  qkv_gemm_kernel<<<gq, 128, 0, stream>>>(xb, wqb, wkb, wvb, bq, bk, bv,
                                          Qs, Kbf, Vt);

  // attention: B*H * (T/32) = 4096 query tiles of 32 rows, 4 waves/block
  int qtiles = (B_ * H_) * (T_ / 32);
  flash_attn_kernel<<<qtiles / 4, 128, 0, stream>>>(Qs, Kbf, Vt, ao);

  // output projection: 2048 tiles of 64x64
  out_proj_kernel<<<((B_ * T_ / 64) * (E_ / 64)) / 4, 128, 0, stream>>>(
      ao, wob, bo, out);
}
